// SequenceMemoryUpdater_72078141162044
// MI455X (gfx1250) — matrix operand order repaired and checked by reading, compiled, not run on hardware
//
#include <hip/hip_runtime.h>
#include <hip/hip_bf16.h>

// ---------------------------------------------------------------------------
// Problem constants (from reference)
// ---------------------------------------------------------------------------
#define N_NODES  1000000
#define N_UPD    100000
#define MSG_DIM  100
#define MEM_DIM  172
#define G3       (3 * MEM_DIM)          // 516 gate columns
#define NTILES   33                     // ceil(516 / 16)
#define KX_STEPS 4                      // ceil(100 / 32)
#define KH_STEPS 6                      // ceil(172 / 32)
#define ROWS     16                     // rows per block (one WMMA M tile)
#define NWAVES   4                      // waves per block
#define BLOCK    (NWAVES * 32)

#define XPAD     128                    // padded K for messages
#define HPAD     192                    // padded K for memory rows
#define NPAD     528                    // padded gate columns (33*16)

#define SUMST    348                    // padded stride for r/z sum region (2*172=344)
#define NST      176                    // padded stride for n region (172)

// shared memory: staging region (16*128 + 16*192 = 5120 floats) aliased with
// gate region (16*348 + 2*16*176 = 11200 floats)
#define SMEM_FLOATS 11200

// pre-swizzled B-fragment counts (one entry = 16 bf16 for one lane)
#define XFRAGS  (NTILES * KX_STEPS * 32)   // 4224
#define HFRAGS  (NTILES * KH_STEPS * 32)   // 6336

typedef __bf16 bf16_t;
typedef bf16_t v16bf __attribute__((ext_vector_type(16)));
typedef float  v8f   __attribute__((ext_vector_type(8)));

__device__ __forceinline__ float sigmoidf_fast(float x) {
    return 1.0f / (1.0f + __expf(-x));
}

// ---------------------------------------------------------------------------
// Prep: swizzle W_ih / W_hh into per-lane bf16 WMMA B-fragments (zero padded),
// and pad the biases. Runs once per launch; tiny (561 KB of weights).
//
// B fragment layout (32x16 K x N, 16-bit): lane holds column N = lane%16;
// lanes 0-15 hold K = kb+[0..15], lanes 16-31 hold K = kb+[16..31].
// Fragment index = (nt * KSTEPS + ks) * 32 + lane; 16 bf16 per fragment.
// ---------------------------------------------------------------------------
__global__ void tgn_prep_kernel(const float* __restrict__ W_ih,
                                const float* __restrict__ W_hh,
                                const float* __restrict__ b_ih,
                                const float* __restrict__ b_hh,
                                bf16_t* __restrict__ wx_frag,   // [XFRAGS][16]
                                bf16_t* __restrict__ wh_frag,   // [HFRAGS][16]
                                float*  __restrict__ bias_pad)  // [2][NPAD]
{
    const int idx = blockIdx.x * blockDim.x + threadIdx.x;

    if (idx < XFRAGS) {
        const int lane = idx & 31;
        const int ks   = (idx >> 5) % KX_STEPS;
        const int nt   = idx / (32 * KX_STEPS);
        const int n    = nt * 16 + (lane & 15);
        const int kb   = ks * 32 + ((lane & 16) ? 16 : 0);
        bf16_t* out = wx_frag + (size_t)idx * 16;
#pragma unroll
        for (int i = 0; i < 16; ++i) {
            const int k = kb + i;
            const float v = (n < G3 && k < MSG_DIM) ? W_ih[(size_t)n * MSG_DIM + k] : 0.0f;
            out[i] = (bf16_t)v;
        }
    } else if (idx < XFRAGS + HFRAGS) {
        const int f    = idx - XFRAGS;
        const int lane = f & 31;
        const int ks   = (f >> 5) % KH_STEPS;
        const int nt   = f / (32 * KH_STEPS);
        const int n    = nt * 16 + (lane & 15);
        const int kb   = ks * 32 + ((lane & 16) ? 16 : 0);
        bf16_t* out = wh_frag + (size_t)f * 16;
#pragma unroll
        for (int i = 0; i < 16; ++i) {
            const int k = kb + i;
            const float v = (n < G3 && k < MEM_DIM) ? W_hh[(size_t)n * MEM_DIM + k] : 0.0f;
            out[i] = (bf16_t)v;
        }
    } else if (idx < XFRAGS + HFRAGS + 2 * NPAD) {
        const int j = idx - (XFRAGS + HFRAGS);
        const int col = (j < NPAD) ? j : j - NPAD;
        const float v = (col < G3) ? ((j < NPAD) ? b_ih[col] : b_hh[col]) : 0.0f;
        bias_pad[j] = v;
    }
}

// ---------------------------------------------------------------------------
// GRU update: gather rows, WMMA the two GEMMs, gate math, scatter results.
// One block = 16 update rows; 4 waves stride over the 33 output N-tiles.
// ---------------------------------------------------------------------------
__global__ __launch_bounds__(BLOCK)
void tgn_gru_wmma_kernel(const int*    __restrict__ node_ids,
                         const float*  __restrict__ messages,
                         const float*  __restrict__ timestamps,
                         const float*  __restrict__ memory,
                         const bf16_t* __restrict__ wx_frag,
                         const bf16_t* __restrict__ wh_frag,
                         const float*  __restrict__ bias_pad,
                         float* __restrict__ out_mem,
                         float* __restrict__ out_lu)
{
    __shared__ float smem[SMEM_FLOATS];
    float* s_x = smem;                       // [16][XPAD]  (staging, aliased)
    float* s_h = smem + ROWS * XPAD;         // [16][HPAD]
    float* s_grz = smem;                     // [16][SUMST] (gates, aliased)
    float* s_in  = smem + ROWS * SUMST;      // [16][NST]
    float* s_hn  = smem + ROWS * SUMST + ROWS * NST;

    const int tid  = threadIdx.x;
    const int lane = tid & 31;
    const int wave = tid >> 5;
    const int mr   = blockIdx.x * ROWS;      // first update row of this tile

    // ---- Stage 0: cooperative zero-padded staging of x / h rows into LDS ----
    // x rows: 100 floats -> 25 float4 of data + 7 float4 of zeros (32 slots/row)
    {
        float4* s_x4 = (float4*)s_x;
        for (int s = tid; s < ROWS * (XPAD / 4); s += BLOCK) {
            const int m = s >> 5;            // XPAD/4 == 32 slots per row
            const int q = s & 31;
            float4 v = make_float4(0.f, 0.f, 0.f, 0.f);
            if (q < MSG_DIM / 4) {
                v = ((const float4*)(messages + (size_t)(mr + m) * MSG_DIM))[q];
            }
            s_x4[s] = v;
        }
        // h rows: 172 floats -> 43 float4 of data + 5 float4 of zeros (48 slots/row)
        float4* s_h4 = (float4*)s_h;
        for (int s = tid; s < ROWS * (HPAD / 4); s += BLOCK) {
            const int m = s / (HPAD / 4);
            const int q = s - m * (HPAD / 4);
            float4 v = make_float4(0.f, 0.f, 0.f, 0.f);
            if (q < MEM_DIM / 4) {
                const int node = node_ids[mr + m];
                v = ((const float4*)(memory + (size_t)node * MEM_DIM))[q];
            }
            s_h4[s] = v;
        }
    }
    __syncthreads();

    // ---- Stage 1: build A fragments from padded LDS (no bounds checks) ----
    // A layout (16x32, M x K): lanes 0-15: row=lane, K=kb+[0..15];
    //                          lanes 16-31: row=lane-16, K=kb+[16..31]
    const int m_a  = lane & 15;
    const int aoff = (lane & 16) ? 16 : 0;

    v16bf ax[KX_STEPS];
#pragma unroll
    for (int ks = 0; ks < KX_STEPS; ++ks) {
        const float* rp = s_x + m_a * XPAD + ks * 32 + aoff;
        v16bf a;
#pragma unroll
        for (int i = 0; i < 8; ++i) {
            a[i]     = (bf16_t)rp[i];
            a[8 + i] = (bf16_t)rp[8 + i];
        }
        ax[ks] = a;
    }
    v16bf ah[KH_STEPS];
#pragma unroll
    for (int ks = 0; ks < KH_STEPS; ++ks) {
        const float* rp = s_h + m_a * HPAD + ks * 32 + aoff;
        v16bf a;
#pragma unroll
        for (int i = 0; i < 8; ++i) {
            a[i]     = (bf16_t)rp[i];
            a[8 + i] = (bf16_t)rp[8 + i];
        }
        ah[ks] = a;
    }
    __syncthreads();   // staging region is now free for gate outputs

    const int mrow = (lane >> 4) * 8;        // C-fragment row base for this lane

    // ---- Stage 2: WMMA over N-tiles (wave-strided); all loads unconditional ----
    const v16bf* wx_v = (const v16bf*)wx_frag;
    const v16bf* wh_v = (const v16bf*)wh_frag;

    for (int nt = wave; nt < NTILES; nt += NWAVES) {
        const int n = nt * 16 + (lane & 15); // this lane's output column (< NPAD)

        const float bi = bias_pad[n];
        const float bh = bias_pad[NPAD + n];
        v8f ci = {bi, bi, bi, bi, bi, bi, bi, bi};
        v8f ch = {bh, bh, bh, bh, bh, bh, bh, bh};

        const v16bf* bx_base = wx_v + (size_t)nt * KX_STEPS * 32 + lane;
        const v16bf* bh_base = wh_v + (size_t)nt * KH_STEPS * 32 + lane;

        // Interleave the independent gi / gh accumulation chains for ILP.
#pragma unroll
        for (int ks = 0; ks < KH_STEPS; ++ks) {
            if (ks < KX_STEPS) {
                v16bf bx = bx_base[ks * 32];
                ci = __builtin_amdgcn_wmma_f32_16x16x32_bf16(
                        false, ax[ks], false, bx, (short)0, ci, false, false);
            }
            v16bf bhf = bh_base[ks * 32];
            ch = __builtin_amdgcn_wmma_f32_16x16x32_bf16(
                    false, ah[ks], false, bhf, (short)0, ch, false, false);
        }

        // Spill C fragments to LDS, pre-combining gi+gh for the r/z regions.
        if (n < G3) {
#pragma unroll
            for (int j = 0; j < 8; ++j) {
                const int m = mrow + j;      // C layout: VGPR j -> row mrow+j
                const float vi = ci[j];
                const float vh = ch[j];
                if (n < 2 * MEM_DIM) {
                    s_grz[m * SUMST + n] = vi + vh;
                } else {
                    const int c = n - 2 * MEM_DIM;
                    s_in[m * NST + c] = vi;
                    s_hn[m * NST + c] = vh;
                }
            }
        }
    }

    __syncthreads();

    // ---- Stage 3: gate math + scatter ----
    for (int e = tid; e < ROWS * MEM_DIM; e += BLOCK) {
        const int m = e / MEM_DIM;
        const int c = e - m * MEM_DIM;
        const int node = node_ids[mr + m];

        const float r  = sigmoidf_fast(s_grz[m * SUMST + c]);
        const float z  = sigmoidf_fast(s_grz[m * SUMST + c + MEM_DIM]);
        const float nn = tanhf(s_in[m * NST + c] + r * s_hn[m * NST + c]);
        const float hold = memory[(size_t)node * MEM_DIM + c];

        out_mem[(size_t)node * MEM_DIM + c] = (1.0f - z) * nn + z * hold;
    }

    if (tid < ROWS) {
        const int node = node_ids[mr + tid];
        out_lu[node] = timestamps[mr + tid];
    }
}

// ---------------------------------------------------------------------------
// Launch: prep weights, copy-through the full tables, scatter the GRU updates.
// ---------------------------------------------------------------------------
extern "C" void kernel_launch(void* const* d_in, const int* in_sizes, int n_in,
                              void* d_out, int out_size, void* d_ws, size_t ws_size,
                              hipStream_t stream) {
    (void)in_sizes; (void)n_in; (void)out_size; (void)ws_size;

    const int*   node_ids   = (const int*)  d_in[0];
    const float* messages   = (const float*)d_in[1];
    const float* timestamps = (const float*)d_in[2];
    const float* memory     = (const float*)d_in[3];
    const float* last_upd   = (const float*)d_in[4];
    const float* W_ih       = (const float*)d_in[5];
    const float* W_hh       = (const float*)d_in[6];
    const float* b_ih       = (const float*)d_in[7];
    const float* b_hh       = (const float*)d_in[8];

    float* out_mem = (float*)d_out;
    float* out_lu  = out_mem + (size_t)N_NODES * MEM_DIM;

    // Workspace partition: swizzled bf16 weight fragments + padded biases.
    bf16_t* wx_frag = (bf16_t*)d_ws;                          // XFRAGS*16 bf16
    bf16_t* wh_frag = wx_frag + (size_t)XFRAGS * 16;          // HFRAGS*16 bf16
    float*  bias_pad = (float*)(wh_frag + (size_t)HFRAGS * 16); // 2*NPAD floats

    // Pass 0: swizzle weights into WMMA fragment layout (tiny).
    {
        const int total = XFRAGS + HFRAGS + 2 * NPAD;
        tgn_prep_kernel<<<(total + 127) / 128, 128, 0, stream>>>(
            W_ih, W_hh, b_ih, b_hh, wx_frag, wh_frag, bias_pad);
    }

    // Pass 1: identity copy of the full memory table + last_update (BW-bound floor).
    hipMemcpyAsync(out_mem, memory, (size_t)N_NODES * MEM_DIM * sizeof(float),
                   hipMemcpyDeviceToDevice, stream);
    hipMemcpyAsync(out_lu, last_upd, (size_t)N_NODES * sizeof(float),
                   hipMemcpyDeviceToDevice, stream);

    // Pass 2: GRU update on the 100k touched rows (overwrites copied rows).
    tgn_gru_wmma_kernel<<<N_UPD / ROWS, BLOCK, 0, stream>>>(
        node_ids, messages, timestamps, memory,
        wx_frag, wh_frag, bias_pad, out_mem, out_lu);
}
